// MOE_Resnet18_37933151158765
// MI455X (gfx1250) — matrix operand order repaired and checked by reading, not compiled, not run on hardware
//
#include <hip/hip_runtime.h>

// MI455X (gfx1250) MoE-ResNet18 forward, bf16 WMMA implicit-GEMM convolutions.
// ~142 GFLOP of conv GEMMs: compute bound -> V_WMMA_F32_16X16X32_BF16.
// 64x64 block tile, 4 wave32/WG, one B fragment reused across 4 WMMAs.
// Weights are pre-converted once per expert to zero-padded bf16 rows
// ([COUT][KW->mult-of-32]) so the hot loop's B path is two aligned b128 loads
// with no conversion math. Requires ws_size >= ~56 MB.

typedef __attribute__((ext_vector_type(16))) __bf16 v16bf;
typedef __attribute__((ext_vector_type(8)))  float  v8f;

struct U16x8 { unsigned short s[8]; };
struct B32B  { uint4 lo, hi; };

__device__ __forceinline__ unsigned short f2bf(float f) {
  unsigned int u = __builtin_bit_cast(unsigned int, f);
  unsigned int r = u + 0x7FFFu + ((u >> 16) & 1u);   // round-to-nearest-even
  return (unsigned short)(r >> 16);
}
__device__ __forceinline__ float bf2f(unsigned short h) {
  unsigned int u = ((unsigned int)h) << 16;
  return __builtin_bit_cast(float, u);
}

// ---------------------------------------------------------------------------
// Conv + folded eval-BN (+ residual) (+ ReLU), implicit GEMM:
//   M = 64*HOUT*WOUT (mult of 64), N = COUT (mult of 64), K = CIN*KS*KS.
// Workgroup = 128 threads (4 wave32). Block tile 64(M) x 64(N).
// ---------------------------------------------------------------------------
template <int CIN, int HIN, int COUT, int HOUT, int KS, int STRIDE, int PAD,
          bool HAS_RES, bool RELU>
__global__ __launch_bounds__(128) void conv_bn_wmma(
    const unsigned short* __restrict__ in,   // bf16 [64,CIN,HIN,HIN]
    const unsigned short* __restrict__ wbf,  // bf16 [COUT,KWP] zero-padded
    const float* __restrict__ bn_b, const float* __restrict__ bn_g,
    const float* __restrict__ bn_m, const float* __restrict__ bn_v,
    const unsigned short* __restrict__ res,  // bf16 [64,COUT,HOUT,HOUT] | null
    unsigned short* __restrict__ out)        // bf16 [64,COUT,HOUT,HOUT]
{
  constexpr int KSQ    = KS * KS;
  constexpr int KW     = CIN * KSQ;
  constexpr int KWP    = (KW + 31) & ~31;
  constexpr int KSTEPS = KWP / 32;
  constexpr bool KPAD  = (KW % 32) != 0;     // only the 3-channel stem
  constexpr int HW2    = HOUT * HOUT;
  constexpr int LOGH   = __builtin_ctz(HOUT);
  __shared__ __align__(16) unsigned short At[64 * 40]; // 64 x 32 bf16 (+pad)

  const int tid  = threadIdx.x;
  const int lane = tid & 31, wave = tid >> 5;
  const int hi = lane >> 4, lm = lane & 15;
  const int m_base = blockIdx.x * 64;
  const int o = blockIdx.y * 64 + wave * 16 + lm;  // this lane's out channel
  const unsigned short* wrow = wbf + (long)o * KWP;

  v8f acc[4];
  #pragma unroll
  for (int t = 0; t < 4; ++t) acc[t] = (v8f){0.f,0.f,0.f,0.f,0.f,0.f,0.f,0.f};

  // Per-thread staging coordinates: ki fixed per K-step, mi = tid>>5 + 4*i.
  const int ki = tid & 31;
  const int mi0 = tid >> 5;

  for (int s = 0; s < KSTEPS; ++s) {
    const int k0 = s * 32;
    // ---- stage 64x32 im2col A tile into LDS (2048 elems / 128 thr) ----
    const int k = k0 + ki;                 // fixed k per thread this step
    const int ci = k / KSQ;                // compile-time-const divisors
    const int r  = k - ci * KSQ;
    const int kh = r / KS, kw = r - (r / KS) * KS;
    __syncthreads();
    #pragma unroll
    for (int i = 0; i < 16; ++i) {
      const int mi = mi0 + i * 4;
      unsigned short val = 0;
      if (!KPAD || k < KW) {
        int m  = m_base + mi;
        int ox = m & (HOUT - 1); m >>= LOGH;
        int oy = m & (HOUT - 1); int bb = m >> LOGH;
        int iy = oy * STRIDE - PAD + kh;
        int ix = ox * STRIDE - PAD + kw;
        if ((unsigned)iy < (unsigned)HIN && (unsigned)ix < (unsigned)HIN)
          val = in[(((long)bb * CIN + ci) * HIN + iy) * HIN + ix];
      }
      At[mi * 40 + ki] = val;
    }
    __syncthreads();

    // ---- B fragment: column N=lm, K k0+8*hi..+7 and k0+16+8*hi..+7 ----
    B32B bfrag;
    const unsigned short* wp = wrow + k0 + hi * 8;
    bfrag.lo = *(const uint4*)wp;
    bfrag.hi = *(const uint4*)(wp + 16);
    v16bf bv = __builtin_bit_cast(v16bf, bfrag);
    if (s + 1 < KSTEPS)                           // gfx1250 global_prefetch
      __builtin_prefetch(wrow + k0 + 32, 0, 3);

    // ---- 4 WMMAs reusing the B fragment ----
    #pragma unroll
    for (int t = 0; t < 4; ++t) {
      B32B af;                                    // ISA 16-bit A layout:
      af.lo = *(const uint4*)&At[(t * 16 + lm) * 40 + hi * 8];       // K 8hi..
      af.hi = *(const uint4*)&At[(t * 16 + lm) * 40 + 16 + hi * 8];  // K 16+8hi..
      v16bf av = __builtin_bit_cast(v16bf, af);
      acc[t] = __builtin_amdgcn_wmma_f32_16x16x32_bf16(
                   false, av, false, bv, (short)0, acc[t], false, false);
    }
  }

  // ---- epilogue: folded BN + residual + ReLU, one b128 store per sub-tile.
  // D layout: lane element rr -> m = (m_base + 16t + 8hi) + rr: 8 consecutive
  // m inside one [b,ch] image block => 8 contiguous bf16, 16B aligned.
  const float scale = bn_g[o] * rsqrtf(bn_v[o] + 1e-5f);
  const float shift = bn_b[o] - bn_m[o] * scale;
  #pragma unroll
  for (int t = 0; t < 4; ++t) {
    const int m0 = m_base + t * 16 + hi * 8;
    const int bb = m0 >> (2 * LOGH);
    const int sp = m0 & (HW2 - 1);
    const long idx0 = ((long)bb * COUT + o) * HW2 + sp;
    uint4 rv;
    if constexpr (HAS_RES) rv = *(const uint4*)(res + idx0);
    const unsigned short* rp = (const unsigned short*)&rv;
    U16x8 pk;
    #pragma unroll
    for (int rr = 0; rr < 8; ++rr) {
      float v = acc[t][rr] * scale + shift;
      if constexpr (HAS_RES) v += bf2f(rp[rr]);
      if constexpr (RELU)    v = fmaxf(v, 0.f);
      pk.s[rr] = f2bf(v);
    }
    *(uint4*)(out + idx0) = __builtin_bit_cast(uint4, pk);
  }
}

// fp32 [CO,KW] -> bf16 [CO,KWP] with zero K-padding.
__global__ void cvt_w_bf16(const float* __restrict__ w,
                           unsigned short* __restrict__ out,
                           int kw, int kwp, int n) {
  int i = blockIdx.x * blockDim.x + threadIdx.x;
  if (i < n) {
    int o = i / kwp, k = i - o * kwp;
    out[i] = (k < kw) ? f2bf(w[(long)o * kw + k]) : (unsigned short)0;
  }
}

// Gate: logits[b,e] = x[b,:].gate_W[:,e] + gate_b[e]; top1 = first argmax.
__global__ __launch_bounds__(128) void gate_top1(const float* __restrict__ x,
                                                 const float* __restrict__ gW,
                                                 const float* __restrict__ gb,
                                                 int* __restrict__ top1) {
  __shared__ float logits[4];
  int b = blockIdx.x, e = threadIdx.x >> 5, lane = threadIdx.x & 31;
  const float* xb = x + (long)b * 3072;
  float s = 0.f;
  for (int i = lane; i < 3072; i += 32) s += xb[i] * gW[i * 4 + e];
  for (int off = 16; off > 0; off >>= 1) s += __shfl_down(s, off);
  if (lane == 0) logits[e] = s + gb[e];
  __syncthreads();
  if (threadIdx.x == 0) {
    int best = 0; float bv = logits[0];
    for (int q = 1; q < 4; ++q) if (logits[q] > bv) { bv = logits[q]; best = q; }
    top1[b] = best;
  }
}

__global__ void cvt_f32_bf16(const float* __restrict__ in,
                             unsigned short* __restrict__ out, int n) {
  int i = blockIdx.x * blockDim.x + threadIdx.x;
  if (i < n) out[i] = f2bf(in[i]);
}

// Global average pool [64,512,4,4] -> [64,512], then FC [512,10] + bias.
__global__ __launch_bounds__(256) void gap_fc(const unsigned short* __restrict__ act,
                                              const float* __restrict__ fw,
                                              const float* __restrict__ fb,
                                              float* __restrict__ out) {
  __shared__ float gap[512];
  int b = blockIdx.x;
  for (int c = threadIdx.x; c < 512; c += 256) {
    const unsigned short* p = act + ((long)b * 512 + c) * 16;
    float s = 0.f;
    #pragma unroll
    for (int i = 0; i < 16; ++i) s += bf2f(p[i]);
    gap[c] = s * (1.f / 16.f);
  }
  __syncthreads();
  if (threadIdx.x < 10) {
    int o = threadIdx.x;
    float s = fb[o];
    for (int c = 0; c < 512; ++c) s += gap[c] * fw[c * 10 + o];
    out[b * 10 + o] = s;
  }
}

__global__ void gather_top1(const float* __restrict__ all_out,
                            const int* __restrict__ top1,
                            float* __restrict__ out) {
  int i = blockIdx.x * blockDim.x + threadIdx.x;
  if (i < 640) {
    int b = i / 10, c = i - b * 10;
    out[i] = all_out[((long)top1[b] * 64 + b) * 10 + c];
  }
}

// ---------------------------------------------------------------------------
// Host. Leaf order = jax pytree flatten (dict keys sorted):
//   0 x, 1 gate_W, 2 gate_b, 3-6 bn1{b,g,m,v}, 7 conv1, 8 fc_b, 9 fc_w, then
//   per block: bn1(4),bn2(4),conv1,conv2[,down_bn(4),down_conv].
// ---------------------------------------------------------------------------
extern "C" void kernel_launch(void* const* d_in, const int* in_sizes, int n_in,
                              void* d_out, int out_size, void* d_ws, size_t ws_size,
                              hipStream_t stream) {
  (void)in_sizes; (void)n_in; (void)out_size; (void)ws_size;
  const float* x  = (const float*)d_in[0];
  const float* gW = (const float*)d_in[1];
  const float* gb = (const float*)d_in[2];

  char* ws = (char*)d_ws;
  size_t off = 0;
  unsigned short* b0 = (unsigned short*)(ws + off); off += (size_t)4194304 * 2;
  unsigned short* b1 = (unsigned short*)(ws + off); off += (size_t)4194304 * 2;
  unsigned short* b2 = (unsigned short*)(ws + off); off += (size_t)4194304 * 2;
  unsigned short* b3 = (unsigned short*)(ws + off); off += (size_t)4194304 * 2;
  unsigned short* in0 = (unsigned short*)(ws + off); off += (size_t)196608 * 2;
  unsigned short* wbuf = (unsigned short*)(ws + off); off += (size_t)11159552 * 2;
  float* all_out = (float*)(ws + off); off += (size_t)2560 * 4;
  int* top1 = (int*)(ws + off); off += 64 * 4;

  gate_top1<<<64, 128, 0, stream>>>(x, gW, gb, top1);
  cvt_f32_bf16<<<(196608 + 255) / 256, 256, 0, stream>>>(x, in0, 196608);

  auto BN = [&](int i, int j, int e, int c) {
    return (const float*)d_in[i + j] + (size_t)e * c;
  };

  // Conv weight table, in launch order.
  struct WDesc { int leaf, co, ci, ks; };
  static const WDesc WD[20] = {
    {7, 64, 3, 3},                                               // stem
    {18, 64, 64, 3},  {19, 64, 64, 3},                           // l0 b0
    {28, 64, 64, 3},  {29, 64, 64, 3},                           // l0 b1
    {38, 128, 64, 3}, {44, 128, 64, 1}, {39, 128, 128, 3},       // l1 b0
    {53, 128, 128, 3},{54, 128, 128, 3},                         // l1 b1
    {63, 256, 128, 3},{69, 256, 128, 1},{64, 256, 256, 3},       // l2 b0
    {78, 256, 256, 3},{79, 256, 256, 3},                         // l2 b1
    {88, 512, 256, 3},{94, 512, 256, 1},{89, 512, 512, 3},       // l3 b0
    {103, 512, 512, 3},{104, 512, 512, 3},                       // l3 b1
  };

#define CONV(CI, HI, CO, HO, KS, STR, PD, HR, RL, IN, WP, BNI, RES, OUT)      \
  conv_bn_wmma<CI, HI, CO, HO, KS, STR, PD, HR, RL>                           \
      <<<dim3((64 * HO * HO) / 64, CO / 64), 128, 0, stream>>>(               \
          IN, WP, BN(BNI, 0, e, CO), BN(BNI, 1, e, CO),                       \
          BN(BNI, 2, e, CO), BN(BNI, 3, e, CO), RES, OUT)

  for (int e = 0; e < 4; ++e) {
    // Pre-convert this expert's conv weights to zero-padded bf16 rows.
    unsigned short* wp[20];
    size_t woff = 0;
    for (int c = 0; c < 20; ++c) {
      int kw  = WD[c].ci * WD[c].ks * WD[c].ks;
      int kwp = (kw + 31) & ~31;
      int n   = WD[c].co * kwp;
      wp[c] = wbuf + woff;
      const float* src = (const float*)d_in[WD[c].leaf]
                         + (size_t)e * WD[c].co * kw;
      cvt_w_bf16<<<(n + 255) / 256, 256, 0, stream>>>(src, wp[c], kw, kwp, n);
      woff += (size_t)n;
    }

    // stem 3->64 @32
    CONV(3, 32, 64, 32, 3, 1, 1, false, true, in0, wp[0], 3, nullptr, b0);
    // layer0 block0
    CONV(64, 32, 64, 32, 3, 1, 1, false, true, b0, wp[1], 10, nullptr, b1);
    CONV(64, 32, 64, 32, 3, 1, 1, true,  true, b1, wp[2], 14, b0, b2);
    // layer0 block1
    CONV(64, 32, 64, 32, 3, 1, 1, false, true, b2, wp[3], 20, nullptr, b3);
    CONV(64, 32, 64, 32, 3, 1, 1, true,  true, b3, wp[4], 24, b2, b0);
    // layer1 block0 (64->128, /2)
    CONV(64, 32, 128, 16, 3, 2, 1, false, true,  b0, wp[5], 30, nullptr, b1);
    CONV(64, 32, 128, 16, 1, 2, 0, false, false, b0, wp[6], 40, nullptr, b3);
    CONV(128, 16, 128, 16, 3, 1, 1, true, true,  b1, wp[7], 34, b3, b2);
    // layer1 block1
    CONV(128, 16, 128, 16, 3, 1, 1, false, true, b2, wp[8], 45, nullptr, b1);
    CONV(128, 16, 128, 16, 3, 1, 1, true,  true, b1, wp[9], 49, b2, b0);
    // layer2 block0 (128->256, /2)
    CONV(128, 16, 256, 8, 3, 2, 1, false, true,  b0, wp[10], 55, nullptr, b1);
    CONV(128, 16, 256, 8, 1, 2, 0, false, false, b0, wp[11], 65, nullptr, b3);
    CONV(256, 8, 256, 8, 3, 1, 1, true, true,    b1, wp[12], 59, b3, b2);
    // layer2 block1
    CONV(256, 8, 256, 8, 3, 1, 1, false, true, b2, wp[13], 70, nullptr, b1);
    CONV(256, 8, 256, 8, 3, 1, 1, true,  true, b1, wp[14], 74, b2, b0);
    // layer3 block0 (256->512, /2)
    CONV(256, 8, 512, 4, 3, 2, 1, false, true,  b0, wp[15], 80, nullptr, b1);
    CONV(256, 8, 512, 4, 1, 2, 0, false, false, b0, wp[16], 90, nullptr, b3);
    CONV(512, 4, 512, 4, 3, 1, 1, true, true,   b1, wp[17], 84, b3, b2);
    // layer3 block1
    CONV(512, 4, 512, 4, 3, 1, 1, false, true, b2, wp[18], 95, nullptr, b1);
    CONV(512, 4, 512, 4, 3, 1, 1, true,  true, b1, wp[19], 99, b2, b0);

    gap_fc<<<64, 256, 0, stream>>>(b0,
                                   (const float*)d_in[9] + (size_t)e * 5120,
                                   (const float*)d_in[8] + (size_t)e * 10,
                                   all_out + (size_t)e * 640);
  }
#undef CONV
  gather_top1<<<5, 128, 0, stream>>>(all_out, top1, (float*)d_out);
}
